// MultiHeadAttention_46643344834745
// MI455X (gfx1250) — matrix operand order repaired and checked by reading, not compiled, not run on hardware
//
#include <hip/hip_runtime.h>

typedef __attribute__((ext_vector_type(16))) _Float16 v16h;
typedef __attribute__((ext_vector_type(8)))  float    v8f;
typedef __attribute__((ext_vector_type(4)))  int      v4i;

#define D_MODEL  1024
#define N_HEADS  16
#define HEAD_DIM 64
#define BATCH    2
#define SEQ_Q    2048
#define SEQ_K    2048
#define NEG_BIG  (-1.0e9f)

// -------- CDNA5 async global->LDS copy (ASYNCcnt path) --------
// Builtin signature (from hipcc diagnostic): param0 = v4i32 addrspace(1)*
typedef __attribute__((address_space(1))) v4i GV4;
typedef __attribute__((address_space(3))) v4i LV4;

__device__ __forceinline__ void async_g2l_16(void* lds, const void* g) {
#if __has_builtin(__builtin_amdgcn_global_load_async_to_lds_b128)
  // LDS flat addresses carry the LDS byte offset in the low 32 bits
  __builtin_amdgcn_global_load_async_to_lds_b128(
      (GV4*)(unsigned long long)g,
      (LV4*)(unsigned)(unsigned long long)lds, 0, 0);
#else
  *reinterpret_cast<float4*>(lds) = *reinterpret_cast<const float4*>(g);
#endif
}

__device__ __forceinline__ void async_wait_all() {
#if __has_builtin(__builtin_amdgcn_global_load_async_to_lds_b128)
#if __has_builtin(__builtin_amdgcn_s_wait_asynccnt)
  __builtin_amdgcn_s_wait_asynccnt(0);
#else
  asm volatile("s_wait_asynccnt 0x0" ::: "memory");
#endif
#endif
}

// -------- fragment helpers (CDNA5 WMMA 16x16x32 f16, wave32) --------

__device__ __forceinline__ v8f wmma_f16(v16h a, v16h b, v8f c) {
  return __builtin_amdgcn_wmma_f32_16x16x32_f16(
      /*neg_a=*/false, a, /*neg_b=*/false, b,
      /*c_mod=*/(short)0, c, /*reuse_a=*/false, /*reuse_b=*/false);
}

union FragU { v16h v; unsigned u[8]; };
union Pack4 { _Float16 h[4]; unsigned long long u; };
union Pack2 { _Float16 h[2]; unsigned u; };

// A-matrix (16xK slice, 16-bit): lane L holds row M=L&15.
// VGPR i, half j -> K = (i&4)<<2 | (L>=16 ? 8:0) | (i&3)<<1 | j  (pairs contiguous)
__device__ __forceinline__ v16h load_a_frag(const _Float16* __restrict__ base,
                                            int m0, int k0, int strideH,
                                            int laneM, int hiK8) {
  FragU f;
#pragma unroll
  for (int i = 0; i < 8; ++i) {
    const int K = ((i & 4) << 2) + hiK8 + ((i & 3) << 1);
    f.u[i] = *reinterpret_cast<const unsigned*>(base + (size_t)(m0 + laneM) * strideH + k0 + K);
  }
  return f.v;
}

// Same A fragment but sourced from an f32 tile (convert+pack at read time).
__device__ __forceinline__ v16h load_a_frag_f32(const float* __restrict__ base,
                                                int m0, int strideF,
                                                int laneM, int hiK8) {
  FragU f;
#pragma unroll
  for (int i = 0; i < 8; ++i) {
    const int K = ((i & 4) << 2) + hiK8 + ((i & 3) << 1);
    const float2 p = *reinterpret_cast<const float2*>(base + (m0 + laneM) * strideF + K);
    Pack2 c; c.h[0] = (_Float16)p.x; c.h[1] = (_Float16)p.y;
    f.u[i] = c.u;
  }
  return f.v;
}

// B-matrix (Kx16 slice, 16-bit): lane L holds col N=L&15.
// VGPR i, half j -> K = (i&4)<<2 | (L>=16 ? 8:0) | (i&3)<<1 | j  (pairs contiguous)
// Source must be N-major ("transposed") in halves with given stride.
__device__ __forceinline__ v16h load_b_frag(const _Float16* __restrict__ base,
                                            int n0, int k0, int strideH,
                                            int laneM, int hiK16) {
  FragU f;
#pragma unroll
  for (int i = 0; i < 8; ++i) {
    const int K = ((i & 4) << 2) + (hiK16 >> 1) + ((i & 3) << 1);
    f.u[i] = *reinterpret_cast<const unsigned*>(base + (size_t)(n0 + laneM) * strideH + k0 + K);
  }
  return f.v;
}

// =====================================================================
// Kernels 1/5: C = A(f32, MxK rm) @ W(f32, KxN rm) + bias
// M=4096, N=1024, K=1024.  Block tile 128x64, wave tile 32x32.
// MODE 0: f16 head-split [B,H,L,Dh]      (Q, K projections)
// MODE 1: f32 row-major [M,N]            (output projection)
// MODE 2: f16 head-split transposed [B,H,Dh,L]  (V projection)
// =====================================================================
template <int MODE>
__global__ __launch_bounds__(256) void gemm_proj_kernel(
    const float* __restrict__ A, const float* __restrict__ W,
    const float* __restrict__ bias, void* __restrict__ outp) {
  __shared__ __align__(16) _Float16 As[128 * 32];  // m-major, stride 32
  __shared__ __align__(16) _Float16 Bs[64 * 32];   // n-major, stride 32

  const int tid   = threadIdx.x;
  const int lane  = tid & 31;
  const int wave  = tid >> 5;
  const int waveM = wave >> 1;
  const int waveN = wave & 1;
  const int laneM = lane & 15;
  const int hiK8  = (lane >> 4) << 3;
  const int hiK16 = (lane >> 4) << 4;
  const int m_blk = blockIdx.y * 128;
  const int n_blk = blockIdx.x * 64;

  v8f acc[2][2] = {};

  for (int k0 = 0; k0 < D_MODEL; k0 += 32) {
    {  // stage A tile (128x32) fp32 -> f16
      const int r   = tid >> 1;
      const int seg = (tid & 1) * 16;
      const float4* ap = reinterpret_cast<const float4*>(
          A + (size_t)(m_blk + r) * D_MODEL + k0 + seg);
#pragma unroll
      for (int c = 0; c < 4; ++c) {
        float4 f = ap[c];
        Pack4 p;
        p.h[0] = (_Float16)f.x; p.h[1] = (_Float16)f.y;
        p.h[2] = (_Float16)f.z; p.h[3] = (_Float16)f.w;
        *reinterpret_cast<unsigned long long*>(&As[r * 32 + seg + c * 4]) = p.u;
      }
    }
    {  // stage W tile transposed (n-major 64x32) fp32 -> f16
      const int kr   = tid >> 3;
      const int nseg = (tid & 7) * 8;
      const float* wp = W + (size_t)(k0 + kr) * D_MODEL + n_blk + nseg;
#pragma unroll
      for (int c = 0; c < 8; ++c)
        Bs[(nseg + c) * 32 + kr] = (_Float16)wp[c];
    }
    __syncthreads();

    v16h aF[2], bF[2];
#pragma unroll
    for (int mi = 0; mi < 2; ++mi)
      aF[mi] = load_a_frag(As, waveM * 32 + mi * 16, 0, 32, laneM, hiK8);
#pragma unroll
    for (int ni = 0; ni < 2; ++ni)
      bF[ni] = load_b_frag(Bs, waveN * 32 + ni * 16, 0, 32, laneM, hiK16);
#pragma unroll
    for (int mi = 0; mi < 2; ++mi)
#pragma unroll
      for (int ni = 0; ni < 2; ++ni)
        acc[mi][ni] = wmma_f16(aF[mi], bF[ni], acc[mi][ni]);
    __syncthreads();
  }

  const int rowHi = (lane >> 4) << 3;
#pragma unroll
  for (int mi = 0; mi < 2; ++mi)
#pragma unroll
    for (int ni = 0; ni < 2; ++ni)
#pragma unroll
      for (int r = 0; r < 8; ++r) {
        const int m = m_blk + waveM * 32 + mi * 16 + r + rowHi;
        const int n = n_blk + waveN * 32 + ni * 16 + laneM;
        const float v = acc[mi][ni][r] + bias[n];
        const int b = m >> 11, l = m & 2047;
        const int h = n >> 6,  d = n & 63;
        if (MODE == 0) {
          reinterpret_cast<_Float16*>(outp)
              [(((size_t)(b * N_HEADS + h)) * SEQ_Q + l) * HEAD_DIM + d] = (_Float16)v;
        } else if (MODE == 2) {
          reinterpret_cast<_Float16*>(outp)
              [(((size_t)(b * N_HEADS + h)) * HEAD_DIM + d) * SEQ_K + l] = (_Float16)v;
        } else {
          reinterpret_cast<float*>(outp)[(size_t)m * D_MODEL + n] = v;
        }
      }
}

// =====================================================================
// Kernel 2: scores[b,h] = 0.125 * Q[b,h] @ K[b,h]^T + mask*NEG_BIG
// M=N=2048, K=64.  Block 128x128, wave tile 32x64.  Direct global frags.
// =====================================================================
__global__ __launch_bounds__(256) void scores_kernel(
    const _Float16* __restrict__ qh, const _Float16* __restrict__ kh,
    const int* __restrict__ mask, float* __restrict__ S) {
  const int tid   = threadIdx.x;
  const int lane  = tid & 31;
  const int wave  = tid >> 5;
  const int waveM = wave >> 1;
  const int waveN = wave & 1;
  const int laneM = lane & 15;
  const int hiK8  = (lane >> 4) << 3;
  const int hiK16 = (lane >> 4) << 4;
  const int bh    = blockIdx.z;
  const int b     = bh >> 4;
  const int m_blk = blockIdx.y * 128;
  const int n_blk = blockIdx.x * 128;

  const _Float16* qb = qh + (size_t)bh * SEQ_Q * HEAD_DIM;  // row-major [l][d]
  const _Float16* kb = kh + (size_t)bh * SEQ_K * HEAD_DIM;  // n-major for K^T
  float* Sb = S + (size_t)bh * SEQ_Q * SEQ_K;

  v8f acc[2][4] = {};
#pragma unroll
  for (int k0 = 0; k0 < HEAD_DIM; k0 += 32) {
    v16h aF[2], bF[4];
#pragma unroll
    for (int mi = 0; mi < 2; ++mi)
      aF[mi] = load_a_frag(qb, m_blk + waveM * 32 + mi * 16, k0, HEAD_DIM, laneM, hiK8);
#pragma unroll
    for (int ni = 0; ni < 4; ++ni)
      bF[ni] = load_b_frag(kb, n_blk + waveN * 64 + ni * 16, k0, HEAD_DIM, laneM, hiK16);
#pragma unroll
    for (int mi = 0; mi < 2; ++mi)
#pragma unroll
      for (int ni = 0; ni < 4; ++ni)
        acc[mi][ni] = wmma_f16(aF[mi], bF[ni], acc[mi][ni]);
  }

  const int rowHi = (lane >> 4) << 3;
#pragma unroll
  for (int mi = 0; mi < 2; ++mi)
#pragma unroll
    for (int ni = 0; ni < 4; ++ni)
#pragma unroll
      for (int r = 0; r < 8; ++r) {
        const int m = m_blk + waveM * 32 + mi * 16 + r + rowHi;
        const int n = n_blk + waveN * 64 + ni * 16 + laneM;
        const float v = acc[mi][ni][r] * 0.125f + (float)mask[b * SEQ_K + n] * NEG_BIG;
        Sb[(size_t)m * SEQ_K + n] = v;
      }
}

// =====================================================================
// Kernel 3: in-place row softmax over last dim (2048), one block per row
// =====================================================================
__global__ __launch_bounds__(256) void softmax_kernel(float* __restrict__ S) {
  __shared__ float red[256];
  const int tid = threadIdx.x;
  float* row = S + (size_t)blockIdx.x * SEQ_K;

  float x[8];
  float mx = -3.402823466e38f;
#pragma unroll
  for (int i = 0; i < 8; ++i) { x[i] = row[i * 256 + tid]; mx = fmaxf(mx, x[i]); }
  red[tid] = mx;
  __syncthreads();
  for (int off = 128; off > 0; off >>= 1) {
    if (tid < off) red[tid] = fmaxf(red[tid], red[tid + off]);
    __syncthreads();
  }
  const float m = red[0];
  __syncthreads();

  float s = 0.f;
#pragma unroll
  for (int i = 0; i < 8; ++i) { x[i] = __expf(x[i] - m); s += x[i]; }
  red[tid] = s;
  __syncthreads();
  for (int off = 128; off > 0; off >>= 1) {
    if (tid < off) red[tid] += red[tid + off];
    __syncthreads();
  }
  const float inv = 1.0f / red[0];
#pragma unroll
  for (int i = 0; i < 8; ++i) row[i * 256 + tid] = x[i] * inv;
}

// =====================================================================
// Kernel 4: attn_out[b,h] = P[b,h] @ V[b,h]
// M=2048, N=64, K=2048.  Block 256x64, wave tile 32x64.
// P tile async-copied raw f32 to LDS (ASYNCcnt), converted at frag build.
// V pre-transposed [B,H,Dh,Lk] -> B-frags straight from global f16.
// =====================================================================
__global__ __launch_bounds__(256) void attnv_kernel(
    const float* __restrict__ S, const _Float16* __restrict__ vt,
    float* __restrict__ out) {
  __shared__ __align__(16) float Ps[256 * 32];  // m-major f32, stride 32 (32 KB)

  const int tid   = threadIdx.x;
  const int lane  = tid & 31;
  const int wave  = tid >> 5;       // 0..7 -> 32 rows each
  const int laneM = lane & 15;
  const int hiK8  = (lane >> 4) << 3;
  const int hiK16 = (lane >> 4) << 4;
  const int bh    = blockIdx.y;
  const int b     = bh >> 4;
  const int h     = bh & 15;
  const int m_blk = blockIdx.x * 256;

  const float*    Sb = S  + (size_t)bh * SEQ_Q * SEQ_K;
  const _Float16* vb = vt + (size_t)bh * HEAD_DIM * SEQ_K;  // d-major [d][l]

  v8f acc[2][4] = {};
  for (int k0 = 0; k0 < SEQ_K; k0 += 32) {
    // ---- async stage of P tile (256x32 f32), one row per thread ----
    const float* src = Sb + (size_t)(m_blk + tid) * SEQ_K + k0;
#pragma unroll
    for (int c = 0; c < 8; ++c)
      async_g2l_16(&Ps[tid * 32 + c * 4], src + c * 4);
    if (k0 + 32 < SEQ_K) __builtin_prefetch(src + 32, 0, 0);  // next K tile
    async_wait_all();
    __syncthreads();

    v16h aF[2], bF[4];
#pragma unroll
    for (int mi = 0; mi < 2; ++mi)
      aF[mi] = load_a_frag_f32(Ps, wave * 32 + mi * 16, 32, laneM, hiK8);
#pragma unroll
    for (int ni = 0; ni < 4; ++ni)
      bF[ni] = load_b_frag(vb, ni * 16, k0, SEQ_K, laneM, hiK16);
#pragma unroll
    for (int mi = 0; mi < 2; ++mi)
#pragma unroll
      for (int ni = 0; ni < 4; ++ni)
        acc[mi][ni] = wmma_f16(aF[mi], bF[ni], acc[mi][ni]);
    __syncthreads();
  }

  const int rowHi = (lane >> 4) << 3;
#pragma unroll
  for (int mi = 0; mi < 2; ++mi)
#pragma unroll
    for (int ni = 0; ni < 4; ++ni)
#pragma unroll
      for (int r = 0; r < 8; ++r) {
        const int l = m_blk + wave * 32 + mi * 16 + r + rowHi;
        const int d = ni * 16 + laneM;
        out[((size_t)(b * SEQ_Q + l)) * D_MODEL + h * HEAD_DIM + d] = acc[mi][ni][r];
      }
}

// =====================================================================
extern "C" void kernel_launch(void* const* d_in, const int* in_sizes, int n_in,
                              void* d_out, int out_size, void* d_ws, size_t ws_size,
                              hipStream_t stream) {
  (void)in_sizes; (void)n_in; (void)out_size; (void)ws_size;
  const float* query = (const float*)d_in[0];
  const float* key_  = (const float*)d_in[1];
  const float* value = (const float*)d_in[2];
  const int*   mask  = (const int*)d_in[3];
  const float* Wq = (const float*)d_in[4];
  const float* bq = (const float*)d_in[5];
  const float* Wk = (const float*)d_in[6];
  const float* bk = (const float*)d_in[7];
  const float* Wv = (const float*)d_in[8];
  const float* bv = (const float*)d_in[9];
  const float* Wo = (const float*)d_in[10];
  const float* bo = (const float*)d_in[11];

  float* out   = (float*)d_out;                              // (B,LQ,D) fp32
  float* attnW = out + (size_t)BATCH * SEQ_Q * D_MODEL;      // (B,H,LQ,LK) fp32

  // workspace: f16 Q/K head-split, f16 V transposed, fp32 attn_out = 40 MB
  const size_t hele = (size_t)BATCH * N_HEADS * SEQ_Q * HEAD_DIM;
  _Float16* qh = (_Float16*)d_ws;
  _Float16* kh = qh + hele;
  _Float16* vt = kh + hele;
  float* attn_out = (float*)(vt + hele);

  dim3 blk(256);
  dim3 gProj(D_MODEL / 64, (BATCH * SEQ_Q) / 128);
  gemm_proj_kernel<0><<<gProj, blk, 0, stream>>>(query, Wq, bq, (void*)qh);
  gemm_proj_kernel<0><<<gProj, blk, 0, stream>>>(key_,  Wk, bk, (void*)kh);
  gemm_proj_kernel<2><<<gProj, blk, 0, stream>>>(value, Wv, bv, (void*)vt);

  dim3 gScores(SEQ_K / 128, SEQ_Q / 128, BATCH * N_HEADS);
  scores_kernel<<<gScores, blk, 0, stream>>>(qh, kh, mask, attnW);

  softmax_kernel<<<dim3(BATCH * N_HEADS * SEQ_Q), blk, 0, stream>>>(attnW);

  dim3 gAV(SEQ_Q / 256, BATCH * N_HEADS);
  attnv_kernel<<<gAV, blk, 0, stream>>>(attnW, vt, attn_out);

  gemm_proj_kernel<1><<<gProj, blk, 0, stream>>>(attn_out, Wo, bo, (void*)out);
}
